// StaticEncEmbedding_23063974380040
// MI455X (gfx1250) — compile-verified, matrix operand order checked by reading, not run on hardware
//
#include <hip/hip_runtime.h>
#include <hip/hip_bf16.h>
#include <math.h>

// ---------------------------------------------------------------------------
// Problem constants (from the reference)
// ---------------------------------------------------------------------------
#define BATCH 16384
#define FN    32
#define FC    8
#define HDIM  64
#define EDIM  128
#define VSZ   1000

#define KNUM   (FN * HDIM)          // 2048
#define KCAT   (FC * HDIM)          // 512
#define KTOT   (KNUM + KCAT)        // 2560
#define CHUNKS (KTOT / 32)          // 80 K-chunks of 32 (64 num + 16 cat)
#define NTILES (EDIM / 16)          // 8 N tiles of 16

#define CHUNK_ELTS (32 * EDIM)      // 4096 bf16 per packed-B chunk (8 KB)
#define NPACK  (KTOT * EDIM)        // 327680 bf16 elements (640 KB)

typedef __attribute__((ext_vector_type(16))) __bf16 v16bf;
typedef __attribute__((ext_vector_type(8)))  float  v8f;

__device__ __forceinline__ float gelu_exact(float x) {
    // Matches jax.nn.gelu(..., approximate=False): 0.5*x*(1+erf(x/sqrt(2)))
    return 0.5f * x * (1.0f + erff(x * 0.70710678118654752f));
}

__device__ __forceinline__ void wait_asynccnt0() {
#if __has_builtin(__builtin_amdgcn_s_wait_asynccnt)
    __builtin_amdgcn_s_wait_asynccnt(0);
#else
    asm volatile("s_wait_asynccnt 0x0" ::: "memory");
#endif
}

// Async DMA: 16 bytes global -> LDS, tracked by ASYNCcnt (no VGPR round-trip).
__device__ __forceinline__ void async_copy_b128(uint32_t lds_byte_addr,
                                                const void* gaddr) {
    asm volatile("global_load_async_to_lds_b128 %0, %1, off"
                 :: "v"(lds_byte_addr), "v"(gaddr)
                 : "memory");
}

// ---------------------------------------------------------------------------
// Pre-pack kernel: reorder W2 (FN,H,E) ++ Wc (FC,H,E) into WMMA B-fragment
// order so the GEMM loop streams B with contiguous 128-bit transfers.
//
// Packed element tid = ((c*8 + t)*32 + lane)*16 + e:
//   K    = 32*c + 8*(lane>>4) + (e&7) + 16*(e>>3)     (ISA 16-bit B 32x16 map)
//   Ncol = 16*t + (lane&15)
// Also folds bias[e] = sum_f b2[f,e] + sum_f bc[f,e] into d_ws after Bpack.
// ---------------------------------------------------------------------------
__global__ void pack_kernel(const float* __restrict__ W2,
                            const float* __restrict__ Wc,
                            const float* __restrict__ b2,
                            const float* __restrict__ bc,
                            __bf16* __restrict__ Bpack,
                            float* __restrict__ bias) {
    int tid = blockIdx.x * blockDim.x + threadIdx.x;
    if (tid < NPACK) {
        int e    = tid & 15;
        int L    = (tid >> 4) & 31;
        int ct   = tid >> 9;           // c*8 + t
        int t    = ct & 7;
        int c    = ct >> 3;
        int half = L >> 4;
        int koff = (half << 3) + (e & 7) + ((e >> 3) << 4);
        int K    = (c << 5) + koff;
        int ecol = (t << 4) + (L & 15);
        float v  = (K < KNUM) ? W2[K * EDIM + ecol]
                              : Wc[(K - KNUM) * EDIM + ecol];
        Bpack[tid] = (__bf16)v;
    } else if (tid < NPACK + EDIM) {
        int j = tid - NPACK;
        float s = 0.0f;
        #pragma unroll
        for (int f = 0; f < FN; ++f) s += b2[f * EDIM + j];
        #pragma unroll
        for (int f = 0; f < FC; ++f) s += bc[f * EDIM + j];
        bias[j] = s;
    }
}

// ---------------------------------------------------------------------------
// Fused GEMM: out[b,e] = sum_K A[b,K]*Bmat[K,e] + bias[e]
//   A[b, K<2048]  = gelu(x_num[b, K/64]*W1[K/64, K%64] + b1[...])  (in-register)
//   A[b, K>=2048] = Etab[f, x_cat[b,f], hh]                        (L2 gather)
// Packed B is staged per-chunk into LDS with global_load_async_to_lds_b128,
// double-buffered: chunk c+1's DMA overlaps chunk c's erf/WMMA work.
// All 8 B fragments are pre-loaded to registers before the WMMA burst so the
// 8 WMMAs issue back-to-back (no per-WMMA s_wait_dscnt round-trip).
// Each wave: 16 rows x 128 cols (8 acc tiles), block = 128 threads = 4 waves.
// ---------------------------------------------------------------------------
__global__ __launch_bounds__(128) void fused_gemm_kernel(
        const float* __restrict__ x_num,
        const int*   __restrict__ x_cat,
        const float* __restrict__ W1,
        const float* __restrict__ b1,
        const float* __restrict__ Etab,
        const __bf16* __restrict__ Bpack,
        const float* __restrict__ bias,
        float* __restrict__ out) {
    __shared__ __align__(16) __bf16 smem[2 * CHUNK_ELTS];   // 2 x 8 KB

    const int tid     = threadIdx.x;
    const int lane    = tid & 31;
    const int wave    = tid >> 5;
    const int half    = lane >> 4;       // which K-octet group this lane holds
    const int l16     = lane & 15;       // A: M row / B,C: N col within tile
    const int rowbase = blockIdx.x * 64 + wave * 16;
    const int arow    = rowbase + l16;   // global row whose A data we hold

    // Cooperative async stage of one packed-B chunk (8 KB) into LDS buffer.
    // 128 threads x 4 rounds x 16 B.
    auto stage = [&](int c, int buf) {
        const __bf16* g = Bpack + (size_t)c * CHUNK_ELTS + tid * 8;
        uint32_t l = (uint32_t)(uintptr_t)(smem + buf * CHUNK_ELTS + tid * 8);
        #pragma unroll
        for (int r = 0; r < 4; ++r) {
            async_copy_b128(l + r * 2048, g + r * 1024);
        }
    };

    v8f acc[NTILES];
    #pragma unroll
    for (int t = 0; t < NTILES; ++t) acc[t] = (v8f)(0.0f);

    stage(0, 0);

    for (int c = 0; c < CHUNKS; ++c) {
        const int buf = c & 1;

        wait_asynccnt0();     // my share of buffer `buf` has landed in LDS
        __syncthreads();      // everyone's share has landed

        if (c + 1 < CHUNKS) stage(c + 1, buf ^ 1);   // overlap next DMA

        // ---- Pre-load all 8 B fragments (16x ds_load_b128, clause-able) ----
        const __bf16* bbase = smem + buf * CHUNK_ELTS + lane * 16;
        v16bf bfrag[NTILES];
        #pragma unroll
        for (int t = 0; t < NTILES; ++t) {
            bfrag[t] = *(const v16bf*)(bbase + t * 512);
        }

        // ---- Build A fragment (16 bf16 per lane, ISA 16-bit 16x32 layout) --
        // (long erf chains overlap the outstanding LDS loads above)
        v16bf af;
        if (c < KNUM / 32) {
            // numeric path: gelu(x*W1 + b1), computed exactly once per element
            const int   f  = c >> 1;
            const float x  = x_num[arow * FN + f];
            const int   hb = ((c & 1) << 5) + (half << 3);
            const float* wp = W1 + f * HDIM + hb;
            const float* bp = b1 + f * HDIM + hb;
            float a[16];
            #pragma unroll
            for (int j = 0; j < 8; ++j) {
                a[j]     = gelu_exact(fmaf(x, wp[j],      bp[j]));
                a[8 + j] = gelu_exact(fmaf(x, wp[16 + j], bp[16 + j]));
            }
            #pragma unroll
            for (int e = 0; e < 16; ++e) af[e] = (__bf16)a[e];
        } else {
            // categorical path: gather embedding row from Etab (L2-resident)
            const int cc  = c - KNUM / 32;
            const int f   = cc >> 1;
            const int idx = x_cat[arow * FC + f];
            const int hb  = ((cc & 1) << 5) + (half << 3);
            const float* ep = Etab + ((size_t)f * VSZ + idx) * HDIM + hb;
            #pragma unroll
            for (int j = 0; j < 8; ++j) {
                af[j]     = (__bf16)ep[j];
                af[8 + j] = (__bf16)ep[16 + j];
            }
        }

        // ---- 8 back-to-back WMMAs (same A, disjoint acc: no RAW hazards) ---
        #pragma unroll
        for (int t = 0; t < NTILES; ++t) {
            acc[t] = __builtin_amdgcn_wmma_f32_16x16x32_bf16(
                false, af, false, bfrag[t], (short)0, acc[t], false, false);
        }

        __syncthreads();      // reads of `buf` done before it is re-staged
    }

    // ---- Epilogue: bias add + store (C/D layout: VGPR r -> M = r + 8*half) --
    #pragma unroll
    for (int t = 0; t < NTILES; ++t) {
        const int ecol = t * 16 + l16;
        const float bv = bias[ecol];
        #pragma unroll
        for (int r = 0; r < 8; ++r) {
            const int grow = rowbase + half * 8 + r;
            out[(size_t)grow * EDIM + ecol] = acc[t][r] + bv;
        }
    }
}

// ---------------------------------------------------------------------------
// Launch: pack B/bias into d_ws, then run the fused WMMA GEMM.
// ---------------------------------------------------------------------------
extern "C" void kernel_launch(void* const* d_in, const int* in_sizes, int n_in,
                              void* d_out, int out_size, void* d_ws, size_t ws_size,
                              hipStream_t stream) {
    const float* x_num = (const float*)d_in[0];
    const int*   x_cat = (const int*)  d_in[1];
    const float* W1    = (const float*)d_in[2];
    const float* b1    = (const float*)d_in[3];
    const float* W2    = (const float*)d_in[4];
    const float* b2    = (const float*)d_in[5];
    const float* Etab  = (const float*)d_in[6];
    const float* Wc    = (const float*)d_in[7];
    const float* bc    = (const float*)d_in[8];
    float* out = (float*)d_out;

    __bf16* Bpack = (__bf16*)d_ws;
    float*  bias  = (float*)((char*)d_ws + (size_t)NPACK * sizeof(__bf16));

    {
        int total  = NPACK + EDIM;
        int blocks = (total + 255) / 256;
        pack_kernel<<<blocks, 256, 0, stream>>>(W2, Wc, b2, bc, Bpack, bias);
    }
    {
        int blocks = BATCH / 64;   // 64 rows per block (4 waves x 16 rows)
        fused_gemm_kernel<<<blocks, 128, 0, stream>>>(
            x_num, x_cat, W1, b1, Etab, Bpack, bias, out);
    }
}
